// RotarySelfAttention_45071386805036
// MI455X (gfx1250) — matrix operand order repaired and checked by reading, not compiled, not run on hardware
//
#include <hip/hip_runtime.h>

typedef __bf16 bf16_t;
typedef __attribute__((ext_vector_type(16))) __bf16 v16bf;
typedef __attribute__((ext_vector_type(8)))  float  v8f;

#define CB 2
#define CT 2048
#define CD 1024
#define CH 16
#define CHD 64

static __device__ __forceinline__ v8f zero8() {
  v8f z;
#pragma unroll
  for (int i = 0; i < 8; ++i) z[i] = 0.0f;
  return z;
}

// DPP16 lane permute (pure VALU, no LDS): CTRL is a DPP control code.
template <int CTRL>
static __device__ __forceinline__ float dpp_movf(float v) {
  int i = __builtin_bit_cast(int, v);
  int r = __builtin_amdgcn_update_dpp(i, i, CTRL, 0xF, 0xF, true);
  return __builtin_bit_cast(float, r);
}

// Butterfly reduction across the 16-lane row group (C-matrix N lanes).
static __device__ __forceinline__ float red16_max(float v) {
  v = fmaxf(v, dpp_movf<0xB1>(v));    // quad_perm xor1
  v = fmaxf(v, dpp_movf<0x4E>(v));    // quad_perm xor2
  v = fmaxf(v, dpp_movf<0x141>(v));   // row_half_mirror
  v = fmaxf(v, dpp_movf<0x140>(v));   // row_mirror
  return v;
}
static __device__ __forceinline__ float red16_sum(float v) {
  v += dpp_movf<0xB1>(v);
  v += dpp_movf<0x4E>(v);
  v += dpp_movf<0x141>(v);
  v += dpp_movf<0x140>(v);
  return v;
}

// A-fragment element->K mapping for 16-bit 16x32 A tiles (ISA 7.12.2).
#define AKIDX(e, h) ((((e) & 8) << 1) + (h) * 8 + ((e) & 7))

// ---------------------------------------------------------------------------
// Kernel 1: QKV projection + RoPE, double-buffered bf16 WMMA GEMM.
// grid = (64, 48), block = 128 (4 waves).
// Q,K out: bf16 [B,H,T,HD]; V out: bf16 [B,H,HD,T] (transposed for PV GEMM).
// ---------------------------------------------------------------------------
__global__ __launch_bounds__(128) void qkv_rope_kernel(
    const float* __restrict__ x,
    const float* __restrict__ wq, const float* __restrict__ wk,
    const float* __restrict__ wv,
    bf16_t* __restrict__ qo, bf16_t* __restrict__ ko, bf16_t* __restrict__ vo)
{
  __shared__ bf16_t lA[2][64 * 40];   // [m][k] pitch 40
  __shared__ bf16_t lB[2][64 * 40];   // transposed [n][k] pitch 40

  const int tid  = threadIdx.x;
  const int lane = tid & 31, wave = tid >> 5;
  const int half = lane >> 4, ln = lane & 15;
  const int mb   = blockIdx.x * 64;
  const int wsel = blockIdx.y >> 4;            // 0=q 1=k 2=v
  const int col0 = (blockIdx.y & 15) * 64;     // head h = col0/64
  const float* w = (wsel == 0) ? wq : (wsel == 1) ? wk : wv;

  float4 fa[4], fb[4];                          // in-flight global data

  auto load_regs = [&](int kc) {
#pragma unroll
    for (int it = 0; it < 4; ++it) {
      int slot = tid + it * 128;
      fa[it] = *(const float4*)(x + (size_t)(mb + (slot >> 3)) * CD + kc + (slot & 7) * 4);
    }
#pragma unroll
    for (int it = 0; it < 4; ++it) {
      int slot = tid + it * 128;
      fb[it] = *(const float4*)(w + (size_t)(kc + (slot >> 4)) * CD + col0 + (slot & 15) * 4);
    }
  };
  auto store_lds = [&](int p) {
#pragma unroll
    for (int it = 0; it < 4; ++it) {
      int slot = tid + it * 128;
      int r = slot >> 3, c4 = (slot & 7) * 4;
      bf16_t* dst = &lA[p][r * 40 + c4];
      dst[0] = (bf16_t)fa[it].x; dst[1] = (bf16_t)fa[it].y;
      dst[2] = (bf16_t)fa[it].z; dst[3] = (bf16_t)fa[it].w;
    }
#pragma unroll
    for (int it = 0; it < 4; ++it) {
      int slot = tid + it * 128;
      int r = slot >> 4, c4 = (slot & 15) * 4;
      lB[p][(c4 + 0) * 40 + r] = (bf16_t)fb[it].x;
      lB[p][(c4 + 1) * 40 + r] = (bf16_t)fb[it].y;
      lB[p][(c4 + 2) * 40 + r] = (bf16_t)fb[it].z;
      lB[p][(c4 + 3) * 40 + r] = (bf16_t)fb[it].w;
    }
  };

  v8f acc[4];
#pragma unroll
  for (int i = 0; i < 4; ++i) acc[i] = zero8();

  load_regs(0);
  store_lds(0);
  __syncthreads();
  int p = 0;
  for (int kc = 0; kc < CD; kc += 32) {
    const bool more = (kc + 32 < CD);
    if (more) load_regs(kc + 32);               // overlap with WMMAs below

    v16bf a;
    {
      const bf16_t* ap = &lA[p][(wave * 16 + ln) * 40];
#pragma unroll
      for (int e = 0; e < 16; ++e) a[e] = ap[half * 16 + e];
    }
#pragma unroll
    for (int nt = 0; nt < 4; ++nt) {
      v16bf bfr;
      const bf16_t* bp = &lB[p][(nt * 16 + ln) * 40 + half * 16];
#pragma unroll
      for (int e = 0; e < 16; ++e) bfr[e] = bp[e];
      acc[nt] = __builtin_amdgcn_wmma_f32_16x16x32_bf16(
          false, a, false, bfr, (short)0, acc[nt], false, false);
    }

    if (more) store_lds(p ^ 1);
    __syncthreads();
    p ^= 1;
  }

  // RoPE (q,k) + store
  const int h = col0 >> 6;
#pragma unroll
  for (int nt = 0; nt < 4; ++nt) {
    const int hd = nt * 16 + ln;
    const float inv_freq = __expf((float)(hd & ~1) * (-0.14391156f)); // -ln(1e4)/64
#pragma unroll
    for (int r = 0; r < 8; ++r) {
      const int mg = mb + wave * 16 + r + 8 * half;
      const int b = mg >> 11, t = mg & (CT - 1);
      float val = acc[nt][r];
      if (wsel < 2) {
        float sv, cv;
        __sincosf((float)t * inv_freq, &sv, &cv);
        float pv = dpp_movf<0xB1>(val);           // partner dim (hd^1)
        float rot = (hd & 1) ? pv : -pv;
        val = val * cv + rot * sv;
      }
      const size_t bh = (size_t)(b * CH + h);
      if (wsel == 0)      qo[(bh * CT + t) * CHD + hd] = (bf16_t)(val * 0.125f);
      else if (wsel == 1) ko[(bh * CT + t) * CHD + hd] = (bf16_t)val;
      else                vo[(bh * CHD + hd) * CT + t] = (bf16_t)val;
    }
  }
}

// ---------------------------------------------------------------------------
// Kernel 2: causal flash attention. grid = (T/64, B*H), block = 128 (4 waves).
// Each wave: 16 query rows, streams keys in chunks of 32.
// ---------------------------------------------------------------------------
__global__ __launch_bounds__(128) void attn_kernel(
    const bf16_t* __restrict__ q, const bf16_t* __restrict__ k,
    const bf16_t* __restrict__ vt, bf16_t* __restrict__ ao)
{
  __shared__ bf16_t plds[4][16 * 34];   // per-wave P tile [m][32 keys], pitch 34

  const int tid = threadIdx.x;
  const int lane = tid & 31, wave = tid >> 5;
  const int half = lane >> 4, ln = lane & 15;
  const int bh = blockIdx.y, b = bh >> 4, h = bh & 15;
  const int q0 = blockIdx.x * 64 + wave * 16;
  const size_t qkb = (size_t)bh * CT * CHD;

  v16bf aq[2];
#pragma unroll
  for (int f = 0; f < 2; ++f) {
    const bf16_t* qp = q + qkb + (size_t)(q0 + ln) * CHD + f * 32;
#pragma unroll
    for (int e = 0; e < 16; ++e) aq[f][e] = qp[half * 16 + e];
  }

  v8f o[4];
#pragma unroll
  for (int f = 0; f < 4; ++f) o[f] = zero8();
  float mrow[8], lrow[8];
#pragma unroll
  for (int r = 0; r < 8; ++r) { mrow[r] = -1e30f; lrow[r] = 0.0f; }

  const int kend = q0 + 16;
  for (int kc = 0; kc < kend; kc += 32) {
    // Hoist ALL K and V fragment loads: 8 contiguous 32B loads in flight,
    // overlapping the QK^T WMMAs and the softmax below.
    v16bf bk[4];                       // [t2*2+f]
#pragma unroll
    for (int t2 = 0; t2 < 2; ++t2)
#pragma unroll
      for (int f = 0; f < 2; ++f) {
        const bf16_t* kp = k + qkb + (size_t)(kc + t2 * 16 + ln) * CHD + f * 32 + half * 16;
#pragma unroll
        for (int e = 0; e < 16; ++e) bk[t2 * 2 + f][e] = kp[e];
      }
    v16bf bv[4];
#pragma unroll
    for (int f = 0; f < 4; ++f) {
      const bf16_t* vp = vt + ((size_t)bh * CHD + f * 16 + ln) * CT + kc + half * 16;
#pragma unroll
      for (int e = 0; e < 16; ++e) bv[f][e] = vp[e];
    }

    v8f s[2];
#pragma unroll
    for (int t2 = 0; t2 < 2; ++t2) {
      v8f sc = zero8();
      sc = __builtin_amdgcn_wmma_f32_16x16x32_bf16(
          false, aq[0], false, bk[t2 * 2 + 0], (short)0, sc, false, false);
      sc = __builtin_amdgcn_wmma_f32_16x16x32_bf16(
          false, aq[1], false, bk[t2 * 2 + 1], (short)0, sc, false, false);
      const int k0 = kc + t2 * 16;
      if (k0 + 15 > q0) {                      // causal boundary tile
        const int keyn = k0 + ln;
#pragma unroll
        for (int r = 0; r < 8; ++r)
          if (keyn > q0 + r + 8 * half) sc[r] = -1e30f;
      }
      s[t2] = sc;
    }

    float alpha[8];
#pragma unroll
    for (int r = 0; r < 8; ++r) {
      float mn = fmaxf(mrow[r], red16_max(fmaxf(s[0][r], s[1][r])));
      alpha[r] = __expf(mrow[r] - mn);
      float p0 = __expf(s[0][r] - mn);
      float p1 = __expf(s[1][r] - mn);
      lrow[r] = lrow[r] * alpha[r] + red16_sum(p0 + p1);
      mrow[r] = mn;
      const int mr = r + 8 * half;
      plds[wave][mr * 34 + ln]      = (bf16_t)p0;
      plds[wave][mr * 34 + 16 + ln] = (bf16_t)p1;
    }
#pragma unroll
    for (int f = 0; f < 4; ++f)
#pragma unroll
      for (int r = 0; r < 8; ++r) o[f][r] *= alpha[r];

    __builtin_amdgcn_wave_barrier();          // keep P stores before reloads

    v16bf pa;
    {
      const bf16_t* pp = &plds[wave][ln * 34];
#pragma unroll
      for (int e = 0; e < 16; ++e) pa[e] = pp[half * 16 + e];
    }
#pragma unroll
    for (int f = 0; f < 4; ++f)
      o[f] = __builtin_amdgcn_wmma_f32_16x16x32_bf16(
          false, pa, false, bv[f], (short)0, o[f], false, false);
  }

#pragma unroll
  for (int r = 0; r < 8; ++r) {
    const float inv = 1.0f / lrow[r];
#pragma unroll
    for (int f = 0; f < 4; ++f) o[f][r] *= inv;
  }
#pragma unroll
  for (int f = 0; f < 4; ++f) {
    const int hd = f * 16 + ln;
#pragma unroll
    for (int r = 0; r < 8; ++r) {
      const int t = q0 + r + 8 * half;
      ao[((size_t)b * CT + t) * CD + h * CHD + hd] = (bf16_t)o[f][r];
    }
  }
}

// ---------------------------------------------------------------------------
// Kernel 3: output projection attn[4096x1024](bf16) * Wo[1024x1024](f32->bf16)
// grid = (64, 16), block = 128 (4 waves), double-buffered.
// ---------------------------------------------------------------------------
__global__ __launch_bounds__(128) void out_proj_kernel(
    const bf16_t* __restrict__ a, const float* __restrict__ wo,
    float* __restrict__ out)
{
  __shared__ bf16_t lA[2][64 * 40];
  __shared__ bf16_t lB[2][64 * 40];   // transposed [n][k]

  const int tid  = threadIdx.x;
  const int lane = tid & 31, wave = tid >> 5;
  const int half = lane >> 4, ln = lane & 15;
  const int mb   = blockIdx.x * 64;
  const int col0 = blockIdx.y * 64;

  uint4 fa0, fa1;
  float4 fb[4];

  auto load_regs = [&](int kc) {
    const bf16_t* src = a + (size_t)(mb + (tid >> 1)) * CD + kc + (tid & 1) * 16;
    fa0 = *(const uint4*)src;
    fa1 = *(const uint4*)(src + 8);
#pragma unroll
    for (int it = 0; it < 4; ++it) {
      int slot = tid + it * 128;
      fb[it] = *(const float4*)(wo + (size_t)(kc + (slot >> 4)) * CD + col0 + (slot & 15) * 4);
    }
  };
  auto store_lds = [&](int p) {
    bf16_t* dst = &lA[p][(tid >> 1) * 40 + (tid & 1) * 16];
    *(uint4*)dst = fa0;
    *(uint4*)(dst + 8) = fa1;
#pragma unroll
    for (int it = 0; it < 4; ++it) {
      int slot = tid + it * 128;
      int r = slot >> 4, c4 = (slot & 15) * 4;
      lB[p][(c4 + 0) * 40 + r] = (bf16_t)fb[it].x;
      lB[p][(c4 + 1) * 40 + r] = (bf16_t)fb[it].y;
      lB[p][(c4 + 2) * 40 + r] = (bf16_t)fb[it].z;
      lB[p][(c4 + 3) * 40 + r] = (bf16_t)fb[it].w;
    }
  };

  v8f acc[4];
#pragma unroll
  for (int i = 0; i < 4; ++i) acc[i] = zero8();

  load_regs(0);
  store_lds(0);
  __syncthreads();
  int p = 0;
  for (int kc = 0; kc < CD; kc += 32) {
    const bool more = (kc + 32 < CD);
    if (more) load_regs(kc + 32);

    v16bf afr;
    {
      const bf16_t* ap = &lA[p][(wave * 16 + ln) * 40];
#pragma unroll
      for (int e = 0; e < 16; ++e) afr[e] = ap[half * 16 + e];
    }
#pragma unroll
    for (int nt = 0; nt < 4; ++nt) {
      v16bf bfr;
      const bf16_t* bp = &lB[p][(nt * 16 + ln) * 40 + half * 16];
#pragma unroll
      for (int e = 0; e < 16; ++e) bfr[e] = bp[e];
      acc[nt] = __builtin_amdgcn_wmma_f32_16x16x32_bf16(
          false, afr, false, bfr, (short)0, acc[nt], false, false);
    }

    if (more) store_lds(p ^ 1);
    __syncthreads();
    p ^= 1;
  }

#pragma unroll
  for (int nt = 0; nt < 4; ++nt)
#pragma unroll
    for (int r = 0; r < 8; ++r)
      out[(size_t)(mb + wave * 16 + r + 8 * half) * CD + col0 + nt * 16 + ln] =
          acc[nt][r];
}

// ---------------------------------------------------------------------------
extern "C" void kernel_launch(void* const* d_in, const int* in_sizes, int n_in,
                              void* d_out, int out_size, void* d_ws, size_t ws_size,
                              hipStream_t stream) {
  const float* x  = (const float*)d_in[0];
  // d_in[1] = mask (causal, hardcoded in kernel)
  const float* wq = (const float*)d_in[2];
  const float* wk = (const float*)d_in[3];
  const float* wv = (const float*)d_in[4];
  const float* wo = (const float*)d_in[5];
  float* out = (float*)d_out;

  const size_t NELEM = (size_t)CB * CH * CT * CHD;   // 4M elems per tensor
  bf16_t* qws = (bf16_t*)d_ws;
  bf16_t* kws = qws + NELEM;
  bf16_t* vws = kws + NELEM;        // stored transposed [B,H,HD,T]
  bf16_t* aws = vws + NELEM;        // attn output [B*T, H*HD]

  qkv_rope_kernel<<<dim3(CB * CT / 64, 48), 128, 0, stream>>>(
      x, wq, wk, wv, qws, kws, vws);
  attn_kernel<<<dim3(CT / 64, CB * CH), 128, 0, stream>>>(qws, kws, vws, aws);
  out_proj_kernel<<<dim3(CB * CT / 64, CD / 64), 128, 0, stream>>>(aws, wo, out);
}
